// RNN_pjax_26895085207579
// MI455X (gfx1250) — matrix operand order repaired and checked by reading, not compiled, not run on hardware
//
#include <hip/hip_runtime.h>
#include <hip/hip_bf16.h>

// ---------------------------------------------------------------------------
// RNN scan on MI455X (gfx1250), wave32 + WMMA bf16.
//
// 155 GFLOP total; weights 4.5MB bf16 (L2-resident, reused 512x); output
// 128MB f32 (HBM ~5.5us @ 23.3TB/s) => latency-bound sequential scan.
// Recurrence is batch-row-local => 4 independent persistent workgroups of
// 16 batch rows (M=16 == WMMA M), 16 waves each, zero inter-WG sync.
// Per step: embed gather -> GEMM1(1024x512) -> GEMM2(512x512)
//           -> GEMM3(1024x1536 = h_new | logits), LDS-exchanged, 4 barriers.
// ---------------------------------------------------------------------------

typedef __attribute__((ext_vector_type(16))) __bf16 v16bf;
typedef __attribute__((ext_vector_type(8)))  float  v8f;

#define T_STEPS 512
#define STATE   512
#define VOCAB   1024
#define WAVES   16            // 512 threads/block: 4 waves/SIMD, 2x VGPR budget

static __device__ __forceinline__ unsigned short f32_to_bf16_rne(float f) {
  unsigned int u = __float_as_uint(f);
  u += 0x7FFFu + ((u >> 16) & 1u);        // round-to-nearest-even
  return (unsigned short)(u >> 16);
}

union ABits { uint4 q[2]; v16bf v; };

// A-fragment (16x32 bf16) from row-major LDS buffer.
// lane<16 : row=lane,    elems 0..7 = K[kb..kb+7],    8..15 = K[kb+16..kb+23]
// lane>=16: row=lane-16, elems 0..7 = K[kb+8..kb+15], 8..15 = K[kb+24..kb+31]
static __device__ __forceinline__ v16bf load_A_frag(const unsigned short* rowbase,
                                                    int kbase, int hi) {
  ABits u;
  const int c = kbase + hi * 8;
  u.q[0] = *(const uint4*)(rowbase + c);
  u.q[1] = *(const uint4*)(rowbase + c + 16);
  return u.v;
}

// B-fragment from pre-packed weights: tile = 32 lanes x 16 bf16 contiguous.
static __device__ __forceinline__ v16bf load_B_frag(const unsigned short* __restrict__ Bp,
                                                    int tile_idx, int lane) {
  return *(const v16bf*)(Bp + (size_t)(tile_idx * 32 + lane) * 16);
}

// One 16x16 output tile: acc = sum_kt A(kt) * B(nt,kt), KT k-tiles of 32.
// unroll 4: keeps it a loop (no mass hoist -> no scratch spills) while giving
// the scheduler 4 iterations of global/LDS load + WMMA overlap.
static __device__ __forceinline__ v8f gemm_acc(const unsigned short* Arow,
                                               const unsigned short* __restrict__ Bp,
                                               int KT, int nt, int lane, int hi) {
  v8f acc;
#pragma unroll
  for (int i = 0; i < 8; ++i) acc[i] = 0.0f;
#pragma unroll 4
  for (int kt = 0; kt < KT; ++kt) {
    v16bf a = load_A_frag(Arow, kt * 32, hi);
    v16bf b = load_B_frag(Bp, nt * KT + kt, lane);
    acc = __builtin_amdgcn_wmma_f32_16x16x32_bf16(
        /*neg_a=*/false, a, /*neg_b=*/false, b,
        /*c_mod=*/(short)0, acc, /*reuse_a=*/false, /*reuse_b=*/false);
  }
  return acc;
}

// ---------------------------------------------------------------------------
// Pre-pass 1: f32 -> bf16 (embedding table, row-major kept).
// ---------------------------------------------------------------------------
__global__ void cvt_f32_bf16(const float* __restrict__ src,
                             unsigned short* __restrict__ dst, int n) {
  int i = blockIdx.x * blockDim.x + threadIdx.x;
  if (i < n) dst[i] = f32_to_bf16_rne(src[i]);
}

// ---------------------------------------------------------------------------
// Pre-pass 2: pack W (KxN, row-major f32) into WMMA-B fragment order:
//   [nt][kt][lane(32)][elem(16)] bf16, so the hot loop loads 32B/lane.
//   lane<16 : elem e<8 -> W[kt*32 +     e][..], e>=8 -> W[kt*32 + 8 + e][nt*16 + lane]
//   lane>=16: elem e<8 -> W[kt*32 + 8 + e][..], e>=8 -> W[kt*32 + 16 + e][nt*16 + lane-16]
// ---------------------------------------------------------------------------
__global__ void pack_wmma_b(const float* __restrict__ W,
                            unsigned short* __restrict__ Wp, int K, int N) {
  int id = blockIdx.x * blockDim.x + threadIdx.x;
  if (id >= K * N) return;
  int e    = id & 15;
  int lane = (id >> 4) & 31;
  int rest = id >> 9;
  int KT   = K >> 5;
  int kt   = rest % KT;
  int nt   = rest / KT;
  int k    = kt * 32 + ((lane < 16) ? 0 : 8) + ((e < 8) ? e : (e + 8));
  int n    = nt * 16 + (lane & 15);
  Wp[id]   = f32_to_bf16_rne(W[k * N + n]);
}

// ---------------------------------------------------------------------------
// Persistent RNN scan. grid = 4 blocks (one 16-row batch slab each),
// block = 512 threads = 16 wave32 waves. LDS: inp[16][1024] + cat[16][1024].
// ---------------------------------------------------------------------------
__global__ __launch_bounds__(WAVES * 32)
void rnn_scan_kernel(const int* __restrict__ x,
                     const unsigned short* __restrict__ embbf,
                     const unsigned short* __restrict__ W0p,
                     const unsigned short* __restrict__ W1p,
                     const unsigned short* __restrict__ Woutp,
                     const float* __restrict__ b0,
                     const float* __restrict__ b1,
                     const float* __restrict__ bout,
                     float* __restrict__ out) {
  __shared__ unsigned short inp[16][1024];   // [ e_t | h ]
  __shared__ unsigned short cat[16][1024];   // [ h0  | h1 ]

  const int tid  = threadIdx.x;
  const int wave = tid >> 5;     // 0..15
  const int lane = tid & 31;
  const int lcol = lane & 15;    // N column / A row within tile
  const int hi   = lane >> 4;    // half-wave select
  const int m0   = blockIdx.x * 16;

  // 512 threads x 32B cooperative LDS fill pattern (16 rows x 512 bf16)
  const int ldr = tid >> 5;           // 0..15
  const int ldc = (tid & 31) * 16;    // 0..496 step 16

  {
    uint4 z = make_uint4(0u, 0u, 0u, 0u);
    *(uint4*)&inp[ldr][512 + ldc]     = z;     // h(-1) = 0
    *(uint4*)&inp[ldr][512 + ldc + 8] = z;
  }
  __syncthreads();

  for (int t = 0; t < T_STEPS; ++t) {
    // ---- embedding gather: inp[:, 0:512] = emb[x[:, t]] ----
    {
      const int token = x[(m0 + ldr) * T_STEPS + t];
      const uint4* src = (const uint4*)(embbf + (size_t)token * STATE + ldc);
      *(uint4*)&inp[ldr][ldc]     = src[0];
      *(uint4*)&inp[ldr][ldc + 8] = src[1];
    }
    __syncthreads();   // prev GEMM3 h_new writes + e_t visible

    // ---- GEMM1: h0 = relu(inp @ W0 + b0) -> cat[:, 0:512] ----
#pragma unroll 1
    for (int j = 0; j < 2; ++j) {
      const int nt = wave + WAVES * j;
      v8f acc = gemm_acc(&inp[lcol][0], W0p, 32, nt, lane, hi);
      const float bias = b0[nt * 16 + lcol];
#pragma unroll
      for (int v = 0; v < 8; ++v) {
        float f = fmaxf(acc[v] + bias, 0.0f);
        cat[hi * 8 + v][nt * 16 + lcol] = f32_to_bf16_rne(f);
      }
    }
    __syncthreads();

    // ---- GEMM2: h1 = relu(h0 @ W1 + b1) -> cat[:, 512:1024] ----
#pragma unroll 1
    for (int j = 0; j < 2; ++j) {
      const int nt = wave + WAVES * j;
      v8f acc = gemm_acc(&cat[lcol][0], W1p, 16, nt, lane, hi);
      const float bias = b1[nt * 16 + lcol];
#pragma unroll
      for (int v = 0; v < 8; ++v) {
        float f = fmaxf(acc[v] + bias, 0.0f);
        cat[hi * 8 + v][512 + nt * 16 + lcol] = f32_to_bf16_rne(f);
      }
    }
    __syncthreads();

    // ---- GEMM3: cell_out = cat @ Wout + bout ; 6 tiles per wave ----
    // tiles nt<32: h_new = relu(cell_out[:, :512]) -> inp[:, 512:]
    // tiles nt>=32: logits -> global out (B,T,VOCAB) f32
#pragma unroll 1
    for (int j = 0; j < 6; ++j) {
      const int nt = wave + WAVES * j;
      v8f acc = gemm_acc(&cat[lcol][0], Woutp, 32, nt, lane, hi);
      const float bias = bout[nt * 16 + lcol];
      if (nt < 32) {
        const int n0 = 512 + nt * 16 + lcol;
#pragma unroll
        for (int v = 0; v < 8; ++v) {
          float f = fmaxf(acc[v] + bias, 0.0f);
          inp[hi * 8 + v][n0] = f32_to_bf16_rne(f);
        }
      } else {
        const int c = nt * 16 + lcol - 512;   // vocab col 0..1023
#pragma unroll
        for (int v = 0; v < 8; ++v) {
          const size_t brow = (size_t)(m0 + hi * 8 + v);
          out[(brow * T_STEPS + t) * VOCAB + c] = acc[v] + bias;
        }
      }
    }
    __syncthreads();   // h_new complete before next step reads it
  }
}

// ---------------------------------------------------------------------------
extern "C" void kernel_launch(void* const* d_in, const int* in_sizes, int n_in,
                              void* d_out, int out_size, void* d_ws, size_t ws_size,
                              hipStream_t stream) {
  const int*   x    = (const int*)  d_in[0];
  const float* emb  = (const float*)d_in[1];
  const float* W0   = (const float*)d_in[2];
  const float* b0   = (const float*)d_in[3];
  const float* W1   = (const float*)d_in[4];
  const float* b1   = (const float*)d_in[5];
  const float* Wout = (const float*)d_in[6];
  const float* bout = (const float*)d_in[7];
  float*       out  = (float*)d_out;

  // workspace layout (bf16 elements): emb | W0pack | W1pack | Woutpack = 5.77MB
  unsigned short* ws    = (unsigned short*)d_ws;
  unsigned short* embbf = ws;
  unsigned short* W0p   = embbf + 1024 * 512;
  unsigned short* W1p   = W0p   + 1024 * 512;
  unsigned short* Woutp = W1p   +  512 * 512;

  cvt_f32_bf16<<<(1024 * 512) / 256, 256, 0, stream>>>(emb, embbf, 1024 * 512);
  pack_wmma_b <<<(1024 * 512) / 256, 256, 0, stream>>>(W0,   W0p,   1024,  512);
  pack_wmma_b <<<( 512 * 512) / 256, 256, 0, stream>>>(W1,   W1p,    512,  512);
  pack_wmma_b <<<(1024 * 1536) / 256, 256, 0, stream>>>(Wout, Woutp, 1024, 1536);

  rnn_scan_kernel<<<4, WAVES * 32, 0, stream>>>(x, embbf, W0p, W1p, Woutp,
                                                b0, b1, bout, out);
}
